// TriangleMultiplication_58548994179792
// MI455X (gfx1250) — matrix operand order repaired and checked by reading, not compiled, not run on hardware
//
#include <hip/hip_runtime.h>
#include <math.h>

typedef __attribute__((ext_vector_type(16))) __bf16 v16bf;
typedef __attribute__((ext_vector_type(8)))  __bf16 v8bf;
typedef __attribute__((ext_vector_type(8)))  float  v8f;

#define LSEQ 512
#define DDIM 128
#define NPOS (LSEQ*LSEQ)          // 262144 positions

// ---- workspace layout (bytes) ----
#define OFF_LEFT   ((size_t)0)                      // bf16 [128][512][512] = 64 MB
#define OFF_RIGHT  ((size_t)67108864)               // bf16 [128][512][512] = 64 MB
#define OFF_GATE   ((size_t)134217728)              // bf16 [N][128]        = 64 MB
#define OFF_OUTT   ((size_t)201326592)              // f32  [128][N]        = 128 MB
#define OFF_W      ((size_t)335544320)              // bf16 6 x [128][128]

__device__ __forceinline__ float sigmoidf_(float x) { return 1.0f / (1.0f + __expf(-x)); }

// A fragment: 16x32 bf16 tile, row-major, K contiguous, leading dim ld (elements).
__device__ __forceinline__ v16bf load_frag_a(const __bf16* p, int ld, int lane) {
    const int m = lane & 15, sel = lane >> 4;
    const __bf16* rp = p + m * ld + 8 * sel;
    v8bf lo = *(const v8bf*)(rp);
    v8bf hi = *(const v8bf*)(rp + 16);
    v16bf a;
#pragma unroll
    for (int i = 0; i < 8; ++i) { a[i] = lo[i]; a[i + 8] = hi[i]; }
    return a;
}

// B fragment for C = A * B^T: source is row-major [n][k] (K contiguous), 16 rows x 32 k.
__device__ __forceinline__ v16bf load_frag_b(const __bf16* p, int ld, int lane) {
    const int n = lane & 15, sel = lane >> 4;
    const __bf16* rp = p + n * ld + 8 * sel;
    v8bf lo = *(const v8bf*)(rp);
    v8bf hi = *(const v8bf*)(rp + 16);
    v16bf b;
#pragma unroll
    for (int i = 0; i < 8; ++i) { b[i] = lo[i]; b[i + 8] = hi[i]; }
    return b;
}

__device__ __forceinline__ v8f wmma_bf16(v16bf a, v16bf b, v8f c) {
    return __builtin_amdgcn_wmma_f32_16x16x32_bf16(false, a, false, b, (short)0, c, false, false);
}

// async memory -> LDS copy of 16B per lane; INST_OFFSET applies to both sides.
__device__ __forceinline__ void async_copy16(unsigned lds_off, const void* gptr) {
    asm volatile("global_load_async_to_lds_b128 %0, %1, off"
                 :: "v"(lds_off), "v"((unsigned long long)(uintptr_t)gptr) : "memory");
}
__device__ __forceinline__ void async_copy16_o16(unsigned lds_off, const void* gptr) {
    asm volatile("global_load_async_to_lds_b128 %0, %1, off offset:16"
                 :: "v"(lds_off), "v"((unsigned long long)(uintptr_t)gptr) : "memory");
}
__device__ __forceinline__ void wait_async0() {
    asm volatile("s_wait_asynccnt 0x0" ::: "memory");
}

// ---------------- kernel 0: weights fp32 -> bf16 ----------------
__global__ void k_convert_weights(const float* w0, const float* w1, const float* w2,
                                  const float* w3, const float* w4, const float* w5,
                                  __bf16* dst) {
    int i = blockIdx.x * blockDim.x + threadIdx.x;
    if (i >= 6 * 16384) return;
    int w = i >> 14, e = i & 16383;
    const float* srcs[6] = {w0, w1, w2, w3, w4, w5};
    dst[i] = (__bf16)srcs[w][e];
}

// ---------------- kernel 1: LN(D) + 5 projections + gating ----------------
__global__ __launch_bounds__(256)
void k_ln_proj(const float* __restrict__ pair,
               const float* __restrict__ ln_g, const float* __restrict__ ln_b,
               const float* __restrict__ lb, const float* __restrict__ lgb,
               const float* __restrict__ rb, const float* __restrict__ rgb,
               const float* __restrict__ gb,
               const __bf16* __restrict__ wts,       // lw,lgw,rw,rgw,gw (bf16, 16384 each)
               __bf16* __restrict__ left_t, __bf16* __restrict__ right_t,
               __bf16* __restrict__ gate) {
    __shared__ float  tf[128 * 132];                 // fp32 staging; reused as bf16 outstage
    __shared__ __bf16 pln[128 * 136];                // normalized rows, bf16
    __bf16* ost = (__bf16*)tf;                       // outstage [h][128+pad]

    const int t = threadIdx.x;
    const int nBase = blockIdx.x * 128;

    // cooperative fp32 tile load (128 rows x 128 d)
#pragma unroll
    for (int it = 0; it < 16; ++it) {
        int idx = t + it * 256;                      // 4096 float4
        int row = idx >> 5, c4 = idx & 31;
        const float4 v = *(const float4*)(pair + (size_t)(nBase + row) * 128 + c4 * 4);
        float* dr = tf + row * 132 + c4 * 4;
        dr[0] = v.x; dr[1] = v.y; dr[2] = v.z; dr[3] = v.w;
    }
    __syncthreads();

    // LayerNorm over D: two threads per row
    {
        int row = t >> 1, part = t & 1;
        const float* rp = tf + row * 132 + part * 64;
        float s1 = 0.f, s2 = 0.f;
#pragma unroll
        for (int i = 0; i < 64; ++i) { float x = rp[i]; s1 += x; s2 += x * x; }
        s1 += __shfl_xor(s1, 1, 32);
        s2 += __shfl_xor(s2, 1, 32);
        float mu  = s1 * (1.0f / 128.0f);
        float var = s2 * (1.0f / 128.0f) - mu * mu;
        float inv = rsqrtf(var + 1e-5f);
        __bf16* op = pln + row * 136 + part * 64;
#pragma unroll
        for (int i = 0; i < 64; ++i) {
            int dc = part * 64 + i;
            op[i] = (__bf16)((rp[i] - mu) * inv * ln_g[dc] + ln_b[dc]);
        }
    }
    __syncthreads();

    const int wave = t >> 5, lane = t & 31;
    const int m0 = wave * 16;
    const int hcol = lane & 15, sel = lane >> 4;

    // hoist A fragments (rows m0..m0+15, all K=128)
    v16bf af[4];
#pragma unroll
    for (int kk = 0; kk < 4; ++kk) af[kk] = load_frag_a(pln + m0 * 136 + kk * 32, 136, lane);

    const __bf16* Wlw  = wts;
    const __bf16* Wlgw = wts + 16384;
    const __bf16* Wrw  = wts + 2 * 16384;
    const __bf16* Wrgw = wts + 3 * 16384;
    const __bf16* Wgw  = wts + 4 * 16384;

    // ---- pass LEFT ----
    for (int nt = 0; nt < 8; ++nt) {
        v8f aL = {}; v8f aG = {};
#pragma unroll
        for (int kk = 0; kk < 4; ++kk) {
            v16bf bL = load_frag_b(Wlw  + nt * 16 * 128 + kk * 32, 128, lane);
            v16bf bG = load_frag_b(Wlgw + nt * 16 * 128 + kk * 32, 128, lane);
            aL = wmma_bf16(af[kk], bL, aL);
            aG = wmma_bf16(af[kk], bG, aG);
        }
        int h = nt * 16 + hcol;
        float bl = lb[h], blg = lgb[h];
#pragma unroll
        for (int r = 0; r < 8; ++r) {
            float v = sigmoidf_(aG[r] + blg) * (aL[r] + bl);
            ost[h * 136 + m0 + r + 8 * sel] = (__bf16)v;
        }
    }
    __syncthreads();
#pragma unroll
    for (int it = 0; it < 16; ++it) {                // 4096 chunks of 4 halves
        int idx = t + it * 256;
        int h = idx >> 5, n4 = idx & 31;
        *(unsigned long long*)(left_t + (size_t)h * NPOS + nBase + n4 * 4) =
            *(const unsigned long long*)(ost + h * 136 + n4 * 4);
    }
    __syncthreads();

    // ---- pass RIGHT (folds 1/L) ----
    for (int nt = 0; nt < 8; ++nt) {
        v8f aR = {}; v8f aG = {};
#pragma unroll
        for (int kk = 0; kk < 4; ++kk) {
            v16bf bR = load_frag_b(Wrw  + nt * 16 * 128 + kk * 32, 128, lane);
            v16bf bG = load_frag_b(Wrgw + nt * 16 * 128 + kk * 32, 128, lane);
            aR = wmma_bf16(af[kk], bR, aR);
            aG = wmma_bf16(af[kk], bG, aG);
        }
        int h = nt * 16 + hcol;
        float br = rb[h], brg = rgb[h];
#pragma unroll
        for (int r = 0; r < 8; ++r) {
            float v = sigmoidf_(aG[r] + brg) * (aR[r] + br) * (1.0f / 512.0f);
            ost[h * 136 + m0 + r + 8 * sel] = (__bf16)v;
        }
    }
    __syncthreads();
#pragma unroll
    for (int it = 0; it < 16; ++it) {
        int idx = t + it * 256;
        int h = idx >> 5, n4 = idx & 31;
        *(unsigned long long*)(right_t + (size_t)h * NPOS + nBase + n4 * 4) =
            *(const unsigned long long*)(ost + h * 136 + n4 * 4);
    }

    // ---- pass GATE (direct [n][d] store) ----
    for (int nt = 0; nt < 8; ++nt) {
        v8f ac = {};
#pragma unroll
        for (int kk = 0; kk < 4; ++kk) {
            v16bf b = load_frag_b(Wgw + nt * 16 * 128 + kk * 32, 128, lane);
            ac = wmma_bf16(af[kk], b, ac);
        }
        int h = nt * 16 + hcol;
        float bg = gb[h];
#pragma unroll
        for (int r = 0; r < 8; ++r) {
            int n = nBase + m0 + r + 8 * sel;
            gate[(size_t)n * 128 + h] = (__bf16)sigmoidf_(ac[r] + bg);
        }
    }
}

// ---------------- kernel 2: triangle einsum, 128 x (512x512x512) GEMMs ----------------
// Async (memory->LDS) double-buffered staging; one barrier per K-step.
__global__ __launch_bounds__(256)
void k_triangle(const __bf16* __restrict__ left_t, const __bf16* __restrict__ right_t,
                float* __restrict__ out_t) {
    __shared__ __bf16 Abuf[2][128 * 40];
    __shared__ __bf16 Bbuf[2][128 * 40];
    const int t = threadIdx.x, wave = t >> 5, lane = t & 31;
    const int bx = blockIdx.x;
    const int d  = bx >> 4;
    const int iB = ((bx >> 2) & 3) * 128;
    const int jB = (bx & 3) * 128;
    const __bf16* Ag = left_t  + (size_t)d * NPOS + (size_t)iB * 512;
    const __bf16* Bg = right_t + (size_t)d * NPOS + (size_t)jB * 512;

    v8f acc[8];
#pragma unroll
    for (int i = 0; i < 8; ++i) acc[i] = (v8f){};

    // per-thread staging slot: row = t/2, 16-half chunk = (t&1)*16
    const int lrow = t >> 1, lcol = (t & 1) * 16;
    const __bf16* gA0 = Ag + (size_t)lrow * 512 + lcol;   // advance 32 halves per k-step
    const __bf16* gB0 = Bg + (size_t)lrow * 512 + lcol;
    // low 32 bits of a flat __shared__ pointer == wave-relative LDS byte offset
    unsigned ldsA[2], ldsB[2];
    ldsA[0] = (unsigned)(uintptr_t)&Abuf[0][lrow * 40 + lcol];
    ldsA[1] = (unsigned)(uintptr_t)&Abuf[1][lrow * 40 + lcol];
    ldsB[0] = (unsigned)(uintptr_t)&Bbuf[0][lrow * 40 + lcol];
    ldsB[1] = (unsigned)(uintptr_t)&Bbuf[1][lrow * 40 + lcol];

    // prologue: fill buffer 0 with K-step 0 (4 async b128 per thread-pair slot)
    async_copy16    (ldsA[0], gA0);
    async_copy16_o16(ldsA[0], gA0);
    async_copy16    (ldsB[0], gB0);
    async_copy16_o16(ldsB[0], gB0);

    for (int kk = 0; kk < 16; ++kk) {
        const int cur = kk & 1;
        wait_async0();            // this wave's async writes for buffer `cur` are in LDS
        __syncthreads();          // all waves' writes done; all reads of `cur^1` retired
        if (kk < 15) {            // prefetch next K-step into the other buffer
            const __bf16* gA = gA0 + (size_t)(kk + 1) * 32;
            const __bf16* gB = gB0 + (size_t)(kk + 1) * 32;
            async_copy16    (ldsA[cur ^ 1], gA);
            async_copy16_o16(ldsA[cur ^ 1], gA);
            async_copy16    (ldsB[cur ^ 1], gB);
            async_copy16_o16(ldsB[cur ^ 1], gB);
        }
        v16bf a = load_frag_a(&Abuf[cur][wave * 16 * 40], 40, lane);
#pragma unroll
        for (int nt = 0; nt < 8; ++nt) {
            v16bf b = load_frag_b(&Bbuf[cur][nt * 16 * 40], 40, lane);
            acc[nt] = wmma_bf16(a, b, acc[nt]);
        }
    }

    const int col = lane & 15, sel = lane >> 4;
    float* od = out_t + (size_t)d * NPOS;
#pragma unroll
    for (int nt = 0; nt < 8; ++nt)
#pragma unroll
        for (int r = 0; r < 8; ++r)
            od[(size_t)(iB + wave * 16 + r + 8 * sel) * 512 + jB + nt * 16 + col] = acc[nt][r];
}

// ---------------- kernel 3: LN(H) + output projection + gate ----------------
__global__ __launch_bounds__(256)
void k_final(const float* __restrict__ out_t,
             const float* __restrict__ lno_g, const float* __restrict__ lno_b,
             const float* __restrict__ ob, const __bf16* __restrict__ owb,
             const __bf16* __restrict__ gate, float* __restrict__ outp) {
    __shared__ float  ot[128 * 129];
    __shared__ __bf16 aln[128 * 136];
    const int t = threadIdx.x;
    const int nBase = blockIdx.x * 128;

    // gather [d][n] tile (coalesced per d-row)
#pragma unroll
    for (int it = 0; it < 64; ++it) {
        int idx = t + it * 256;
        int d = idx >> 7, n = idx & 127;
        ot[d * 129 + n] = out_t[(size_t)d * NPOS + nBase + n];
    }
    __syncthreads();

    // LayerNorm over H: two threads per position
    {
        int n = t >> 1, part = t & 1;
        float s1 = 0.f, s2 = 0.f;
#pragma unroll
        for (int i = 0; i < 64; ++i) { float x = ot[(part * 64 + i) * 129 + n]; s1 += x; s2 += x * x; }
        s1 += __shfl_xor(s1, 1, 32);
        s2 += __shfl_xor(s2, 1, 32);
        float mu  = s1 * (1.0f / 128.0f);
        float var = s2 * (1.0f / 128.0f) - mu * mu;
        float inv = rsqrtf(var + 1e-5f);
#pragma unroll
        for (int i = 0; i < 64; ++i) {
            int d = part * 64 + i;
            aln[n * 136 + d] = (__bf16)((ot[d * 129 + n] - mu) * inv * lno_g[d] + lno_b[d]);
        }
    }
    __syncthreads();

    const int wave = t >> 5, lane = t & 31, m0 = wave * 16;
    const int col = lane & 15, sel = lane >> 4;
    v16bf af[4];
#pragma unroll
    for (int kk = 0; kk < 4; ++kk) af[kk] = load_frag_a(aln + m0 * 136 + kk * 32, 136, lane);

    for (int nt = 0; nt < 8; ++nt) {
        v8f ac = {};
#pragma unroll
        for (int kk = 0; kk < 4; ++kk) {
            v16bf b = load_frag_b(owb + nt * 16 * 128 + kk * 32, 128, lane);
            ac = wmma_bf16(af[kk], b, ac);
        }
        int dD = nt * 16 + col;
        float bo = ob[dD];
#pragma unroll
        for (int r = 0; r < 8; ++r) {
            int n = nBase + m0 + r + 8 * sel;
            float g = (float)gate[(size_t)n * 128 + dD];
            outp[(size_t)n * 128 + dD] = g * (ac[r] + bo);
        }
    }
}

extern "C" void kernel_launch(void* const* d_in, const int* in_sizes, int n_in,
                              void* d_out, int out_size, void* d_ws, size_t ws_size,
                              hipStream_t stream) {
    const float* pair  = (const float*)d_in[0];
    const float* ln_g  = (const float*)d_in[1];
    const float* ln_b  = (const float*)d_in[2];
    const float* lw    = (const float*)d_in[3];
    const float* lb    = (const float*)d_in[4];
    const float* lgw   = (const float*)d_in[5];
    const float* lgb   = (const float*)d_in[6];
    const float* rw    = (const float*)d_in[7];
    const float* rb    = (const float*)d_in[8];
    const float* rgw   = (const float*)d_in[9];
    const float* rgb   = (const float*)d_in[10];
    const float* gw    = (const float*)d_in[11];
    const float* gb    = (const float*)d_in[12];
    const float* lno_g = (const float*)d_in[13];
    const float* lno_b = (const float*)d_in[14];
    const float* ow    = (const float*)d_in[15];
    const float* ob    = (const float*)d_in[16];

    char* ws = (char*)d_ws;
    __bf16* left_t  = (__bf16*)(ws + OFF_LEFT);
    __bf16* right_t = (__bf16*)(ws + OFF_RIGHT);
    __bf16* gatep   = (__bf16*)(ws + OFF_GATE);
    float*  out_t   = (float*)(ws + OFF_OUTT);
    __bf16* wts     = (__bf16*)(ws + OFF_W);

    k_convert_weights<<<384, 256, 0, stream>>>(lw, lgw, rw, rgw, gw, ow, wts);
    k_ln_proj<<<NPOS / 128, 256, 0, stream>>>(pair, ln_g, ln_b, lb, lgb, rb, rgb, gb,
                                              wts, left_t, right_t, gatep);
    k_triangle<<<128 * 16, 256, 0, stream>>>(left_t, right_t, out_t);
    k_final<<<NPOS / 128, 256, 0, stream>>>(out_t, lno_g, lno_b, ob, wts + 5 * 16384,
                                            gatep, (float*)d_out);
}
